// Qwen3MoeSparseMoeBlock_34720515620976
// MI455X (gfx1250) — matrix operand order repaired and checked by reading, not compiled, not run on hardware
//
#include <hip/hip_runtime.h>
#include <hip/hip_bf16.h>
#include <math.h>

typedef __bf16 bf16_t;
typedef __attribute__((ext_vector_type(16))) __bf16 v16bf;
typedef __attribute__((ext_vector_type(8)))  float  v8f;
typedef uint32_t u32x4 __attribute__((ext_vector_type(4)));
typedef uint32_t u32x8 __attribute__((ext_vector_type(8)));

#define E_ 8
#define H_ 2048
#define I_ 768
#define T_ 4096

#define TM 64      // tokens per block tile
#define TN 64      // output cols per block tile
#define KC 32      // K chunk (matches wmma K)
#define LDA 40     // padded LDS row stride for A tile  [token][k]   (bf16)
#define LDBT 40    // padded LDS row stride for B tile  [n][k]       (bf16)

static __device__ __forceinline__ bf16_t f2bf(float f) {
    return (bf16_t)f;
}

union Pack8 {                 // 8 bf16 packed for one 16B LDS store
    bf16_t h[8];
    uint4  q;
};

union Frag16 {                // 16 bf16 = one WMMA operand fragment (8 VGPRs)
    v16bf v;
    uint4 q[2];
};

// ---------------------------------------------------------------------------
// Kernel 1: router logits + softmax top-2 + normalized combine weights.
// One wave32 per token. comb[e][t] = weight if expert selected else 0.
// ---------------------------------------------------------------------------
__global__ __launch_bounds__(256) void moe_router(
    const float* __restrict__ x, const float* __restrict__ wr,
    float* __restrict__ logits_out, float* __restrict__ comb) {
    const int lane = threadIdx.x & 31;
    const int wave = threadIdx.x >> 5;
    const int t = blockIdx.x * 8 + wave;

    const float* xr = x + (size_t)t * H_;
    float acc[E_];
#pragma unroll
    for (int e = 0; e < E_; ++e) acc[e] = 0.0f;

    for (int h = lane; h < H_; h += 32) {
        float xv = xr[h];
#pragma unroll
        for (int e = 0; e < E_; ++e) acc[e] += xv * wr[h * E_ + e];
    }
#pragma unroll
    for (int e = 0; e < E_; ++e) {
#pragma unroll
        for (int off = 16; off > 0; off >>= 1)
            acc[e] += __shfl_xor(acc[e], off, 32);
    }
    if (lane < E_) {
        logits_out[(size_t)t * E_ + lane] = acc[lane];
        int i1 = 0; float l1 = acc[0];
#pragma unroll
        for (int e = 1; e < E_; ++e) if (acc[e] > l1) { l1 = acc[e]; i1 = e; }
        int i2 = -1; float l2 = -3.0e38f;
#pragma unroll
        for (int e = 0; e < E_; ++e)
            if (e != i1 && acc[e] > l2) { l2 = acc[e]; i2 = e; }
        float p2 = __expf(l2 - l1);
        float inv = 1.0f / (1.0f + p2);
        float w = (lane == i1) ? inv : ((lane == i2) ? p2 * inv : 0.0f);
        comb[(size_t)lane * T_ + t] = w;
    }
}

// ---------------------------------------------------------------------------
// WMMA fragment loads (ISA 7.12.2 layouts, wave32): two contiguous 16B LDS
// vectors per lane -> ds_load_b128 pairs.
// ---------------------------------------------------------------------------
static __device__ __forceinline__ v16bf load_a_frag(const bf16_t* As, int row0,
                                                    int lane) {
    const int row = row0 + (lane & 15);
    const int lg = lane >> 4;
    Frag16 f;
    f.q[0] = *(const uint4*)(As + row * LDA + lg * 8);
    f.q[1] = *(const uint4*)(As + row * LDA + 16 + lg * 8);
    return f.v;
}

static __device__ __forceinline__ v16bf load_b_frag(const bf16_t* Bt, int col0,
                                                    int lane) {
    const int n = col0 + (lane & 15);
    const int kh = (lane >> 4) * 8;
    Frag16 f;
    f.q[0] = *(const uint4*)(Bt + n * LDBT + kh);
    f.q[1] = *(const uint4*)(Bt + n * LDBT + 16 + kh);
    return f.v;
}

static __device__ __forceinline__ void wait_tensorcnt0() {
#if __has_builtin(__builtin_amdgcn_s_wait_tensorcnt)
    __builtin_amdgcn_s_wait_tensorcnt(0);
#else
    asm volatile("s_wait_tensorcnt 0x0" ::: "memory");
#endif
}

// ---------------------------------------------------------------------------
// Kernel 2: per expert, G = X*wi0, U = X*wi1 over 64x64 tiles, K over H.
// Epilogue: hbuf = comb * silu(G) * U   (bf16)
// grid = (T/TM, I/TN, E), block = 256 (8 waves)
// ---------------------------------------------------------------------------
__global__ __launch_bounds__(256) void moe_gateup(
    const float* __restrict__ x, const float* __restrict__ wi0,
    const float* __restrict__ wi1, const float* __restrict__ comb,
    bf16_t* __restrict__ hbuf) {
    __shared__ __align__(16) bf16_t As[TM * LDA];
    __shared__ __align__(16) bf16_t Bt0[TN * LDBT];
    __shared__ __align__(16) bf16_t Bt1[TN * LDBT];
    __shared__ float combs[TM];

    const int e  = blockIdx.z;
    const int t0 = blockIdx.x * TM;
    const int n0 = blockIdx.y * TN;
    const int tid  = threadIdx.x;
    const int lane = tid & 31;
    const int wave = tid >> 5;
    const int mb = wave & 3;
    const int np = wave >> 2;

    const float* w0e = wi0 + (size_t)e * H_ * I_;
    const float* w1e = wi1 + (size_t)e * H_ * I_;

    if (tid < TM) combs[tid] = comb[(size_t)e * T_ + t0 + tid];

    const int ar = tid >> 2;            // A: token row 0..63
    const int ac = (tid & 3) * 8;       // A: k offset 0/8/16/24
    const int bn = tid & 63;            // B: n within tile 0..63
    const int bk = (tid >> 6) * 8;      // B: k base 0/8/16/24

    const float* ax  = x + (size_t)(t0 + ar) * H_ + ac;
    const float* bw0 = w0e + (size_t)bk * I_ + n0 + bn;
    const float* bw1 = w1e + (size_t)bk * I_ + n0 + bn;

    v8f gacc0 = {0,0,0,0,0,0,0,0}, gacc1 = {0,0,0,0,0,0,0,0};
    v8f uacc0 = {0,0,0,0,0,0,0,0}, uacc1 = {0,0,0,0,0,0,0,0};

    for (int k0 = 0; k0 < H_; k0 += KC) {
        // stage X tile [64 tok x 32 k] f32 -> bf16, one b128 store per thread
        {
            float4 f0 = *(const float4*)(ax + k0);
            float4 f1 = *(const float4*)(ax + k0 + 4);
            Pack8 p;
            p.h[0] = f2bf(f0.x); p.h[1] = f2bf(f0.y);
            p.h[2] = f2bf(f0.z); p.h[3] = f2bf(f0.w);
            p.h[4] = f2bf(f1.x); p.h[5] = f2bf(f1.y);
            p.h[6] = f2bf(f1.z); p.h[7] = f2bf(f1.w);
            *(uint4*)&As[ar * LDA + ac] = p.q;
        }
        // stage wi0/wi1 tiles transposed: Bt[n][k], one b128 store each
        {
            const float* s0 = bw0 + (size_t)k0 * I_;
            const float* s1 = bw1 + (size_t)k0 * I_;
            Pack8 p0, p1;
#pragma unroll
            for (int j = 0; j < 8; ++j) {
                p0.h[j] = f2bf(s0[(size_t)j * I_]);
                p1.h[j] = f2bf(s1[(size_t)j * I_]);
            }
            *(uint4*)&Bt0[bn * LDBT + bk] = p0.q;
            *(uint4*)&Bt1[bn * LDBT + bk] = p1.q;
        }
        if (k0 + KC < H_) {
            __builtin_prefetch(ax + k0 + KC, 0, 0);
            __builtin_prefetch(bw0 + (size_t)(k0 + KC) * I_, 0, 0);
            __builtin_prefetch(bw1 + (size_t)(k0 + KC) * I_, 0, 0);
        }
        __syncthreads();

        v16bf a   = load_a_frag(As, mb * 16, lane);
        v16bf b00 = load_b_frag(Bt0, np * 32, lane);
        v16bf b01 = load_b_frag(Bt0, np * 32 + 16, lane);
        v16bf b10 = load_b_frag(Bt1, np * 32, lane);
        v16bf b11 = load_b_frag(Bt1, np * 32 + 16, lane);

        gacc0 = __builtin_amdgcn_wmma_f32_16x16x32_bf16(false, a, false, b00,
                                                        (short)0, gacc0, false, false);
        gacc1 = __builtin_amdgcn_wmma_f32_16x16x32_bf16(false, a, false, b01,
                                                        (short)0, gacc1, false, false);
        uacc0 = __builtin_amdgcn_wmma_f32_16x16x32_bf16(false, a, false, b10,
                                                        (short)0, uacc0, false, false);
        uacc1 = __builtin_amdgcn_wmma_f32_16x16x32_bf16(false, a, false, b11,
                                                        (short)0, uacc1, false, false);
        __syncthreads();
    }

    const int lg = lane >> 4;
    const int nn = lane & 15;
#pragma unroll
    for (int v = 0; v < 8; ++v) {
        const int m = mb * 16 + lg * 8 + v;
        const int t = t0 + m;
        const float cw = combs[m];
        bf16_t* dst = hbuf + ((size_t)e * T_ + t) * I_ + n0 + np * 32 + nn;
        float g0 = gacc0[v], u0 = uacc0[v];
        dst[0]  = f2bf(cw * u0 * g0 / (1.0f + __expf(-g0)));
        float g1 = gacc1[v], u1 = uacc1[v];
        dst[16] = f2bf(cw * u1 * g1 / (1.0f + __expf(-g1)));
    }
}

// ---------------------------------------------------------------------------
// Kernel 3: out[t][h] = sum_e sum_i hbuf[e][t][i] * wo[e][i][h]
// grid = (T/TM, H/TN), block = 256 (8 waves).
// A tile (bf16, layout-exact) is staged by the Tensor Data Mover:
// one tensor_load_to_lds per chunk, with TDM LDS padding reproducing LDA=40
// (row = 32 bf16 = 16 dwords -> pad_interval=3(16dw), pad_amount=3(4dw)).
// ---------------------------------------------------------------------------
__global__ __launch_bounds__(256) void moe_down(
    const bf16_t* __restrict__ hbuf, const float* __restrict__ wo,
    float* __restrict__ out) {
    __shared__ __align__(16) bf16_t As[TM * LDA];
    __shared__ __align__(16) bf16_t Bt[TN * LDBT];

    const int t0 = blockIdx.x * TM;
    const int n0 = blockIdx.y * TN;
    const int tid  = threadIdx.x;
    const int lane = tid & 31;
    const int wave = tid >> 5;
    const int mb = wave & 3;
    const int np = wave >> 2;

    const int bn = tid & 63;
    const int bk = (tid >> 6) * 8;

    // TDM D# group1 (uniform across chunks):
    //  data_size=1 (2B), pad_enable, pad_interval=3 (16 dw), pad_amount=3 (4 dw)
    //  tensor_dim0 = I_ (bits 79:48), tensor_dim1 = E_*T_ (bits 111:80)
    //  tile_dim0 = KC (bits 127:112), tile_dim1 = TM (bits 143:128)
    //  tensor_dim0_stride = I_ (bits 207:160)
    const u32x8 tdm_g1 = {
        (1u << 16) | (1u << 20) | (3u << 22) | (3u << 25),  // w0
        ((uint32_t)I_ & 0xFFFFu) << 16,                     // w1: dim0 lo16
        ((uint32_t)(E_ * T_) & 0xFFFFu) << 16,              // w2: dim1 lo16
        (uint32_t)KC << 16,                                 // w3: tile_dim0
        (uint32_t)TM,                                       // w4: tile_dim1
        (uint32_t)I_,                                       // w5: dim0 stride lo32
        0u, 0u
    };
    const uint32_t lds_a = (uint32_t)(uintptr_t)&As[0];

    v8f acc0 = {0,0,0,0,0,0,0,0}, acc1 = {0,0,0,0,0,0,0,0};

    for (int e = 0; e < E_; ++e) {
        const float* bw = wo + ((size_t)e * I_ + bk) * H_ + n0 + bn;
        for (int k0 = 0; k0 < I_; k0 += KC) {
            // A tile [64 x 32] bf16 via Tensor Data Mover (wave 0 issues)
            if (wave == 0) {
                uint64_t ga = (uint64_t)(uintptr_t)hbuf +
                              (((uint64_t)e * T_ + t0) * I_ + k0) * 2u;
                u32x4 g0 = { 1u,                     // count=1, user descriptor
                             lds_a,                  // lds_addr
                             (uint32_t)ga,           // global_addr[31:0]
                             (uint32_t)((ga >> 32) & 0x01FFFFFFu) | (2u << 30) };
                asm volatile("tensor_load_to_lds %0, %1"
                             :: "s"(g0), "s"(tdm_g1) : "memory");
            }
            // stage B transposed: Bt[n][k] from wo[k][n] (all 256 threads)
            {
                const float* s = bw + (size_t)k0 * H_;
                Pack8 p;
#pragma unroll
                for (int j = 0; j < 8; ++j) p.h[j] = f2bf(s[(size_t)j * H_]);
                *(uint4*)&Bt[bn * LDBT + bk] = p.q;
            }
            if (k0 + KC < I_) {
                __builtin_prefetch(bw + (size_t)(k0 + KC) * H_, 0, 0);
            }
            if (wave == 0) wait_tensorcnt0();
            __syncthreads();

            v16bf a  = load_a_frag(As, mb * 16, lane);
            v16bf b0 = load_b_frag(Bt, np * 32, lane);
            v16bf b1 = load_b_frag(Bt, np * 32 + 16, lane);
            acc0 = __builtin_amdgcn_wmma_f32_16x16x32_bf16(false, a, false, b0,
                                                           (short)0, acc0, false, false);
            acc1 = __builtin_amdgcn_wmma_f32_16x16x32_bf16(false, a, false, b1,
                                                           (short)0, acc1, false, false);
            __syncthreads();
        }
    }

    const int lg = lane >> 4;
    const int nn = lane & 15;
#pragma unroll
    for (int v = 0; v < 8; ++v) {
        const int t = t0 + mb * 16 + lg * 8 + v;
        float* dst = out + (size_t)t * H_ + n0 + np * 32 + nn;
        dst[0]  = acc0[v];
        dst[16] = acc1[v];
    }
}

// ---------------------------------------------------------------------------
extern "C" void kernel_launch(void* const* d_in, const int* in_sizes, int n_in,
                              void* d_out, int out_size, void* d_ws, size_t ws_size,
                              hipStream_t stream) {
    (void)in_sizes; (void)n_in; (void)out_size; (void)ws_size;
    const float* x   = (const float*)d_in[0];   // [B,S,H]
    const float* wr  = (const float*)d_in[1];   // [H,E]
    const float* wi0 = (const float*)d_in[2];   // [E,H,I]
    const float* wi1 = (const float*)d_in[3];   // [E,H,I]
    const float* wo  = (const float*)d_in[4];   // [E,I,H]

    float* out    = (float*)d_out;              // [T,H]
    float* logits = out + (size_t)T_ * H_;      // [T,E]

    float*  comb = (float*)d_ws;                                   // [E,T]
    bf16_t* hbuf = (bf16_t*)((char*)d_ws + (size_t)E_ * T_ * 4);   // [E,T,I]

    moe_router<<<T_ / 8, 256, 0, stream>>>(x, wr, logits, comb);

    dim3 g2(T_ / TM, I_ / TN, E_);
    moe_gateup<<<g2, 256, 0, stream>>>(x, wi0, wi1, comb, hbuf);

    dim3 g3(T_ / TM, H_ / TN);
    moe_down<<<g3, 256, 0, stream>>>(hbuf, wo, out);
}
